// TransformerEncoder_36885179138487
// MI455X (gfx1250) — compile-verified, matrix operand order checked by reading, not run on hardware
//
#include <hip/hip_runtime.h>

// ---------------------------------------------------------------------------
// Transformer encoder forward for MI455X (gfx1250), wave32, WMMA bf16.
// B=16 L=512 D=1024 H=16 HD=64 FF=4096 NL=6.  ~1.36 TFLOP of GEMM work ->
// compute bound on v_wmma_f32_16x16x32_bf16; all GEMM inputs are bf16,
// accumulation f32, softmax/LN in f32.  Activation tiles move to LDS via
// GLOBAL_LOAD_ASYNC_TO_LDS_B128 (ASYNCcnt) when the toolchain exposes it.
// ---------------------------------------------------------------------------

typedef __attribute__((ext_vector_type(16))) __bf16 v16bf;
typedef __attribute__((ext_vector_type(8)))  float  v8f;
typedef int v4i __attribute__((vector_size(16)));        // matches builtin proto
typedef __attribute__((address_space(1))) v4i gv4i_t;    // global int4
typedef __attribute__((address_space(3))) v4i lv4i_t;    // LDS int4 (32-bit ptr)

#define MTOT   8192      // B*L
#define DMODEL 1024
#define SEQ    512
#define NHEAD  16
#define HEADD  64
#define FFDIM  4096
#define NLAYER 6
#define INCAT  1280      // IN_DIM + POS_DIM

#if __has_builtin(__builtin_amdgcn_global_load_async_to_lds_b128)
#define HAVE_ASYNC_LDS 1
#else
#define HAVE_ASYNC_LDS 0
#endif

// 16B global -> LDS copy; async (ASYNCcnt) when available.
// AS3 pointer = low 32 bits of the flat address.
__device__ __forceinline__ void ld16_to_lds(const unsigned short* g,
                                            unsigned short* l) {
#if HAVE_ASYNC_LDS
  __builtin_amdgcn_global_load_async_to_lds_b128(
      (gv4i_t*)(unsigned long long)g,
      (lv4i_t*)(unsigned int)(unsigned long long)l, 0, 0);
#else
  *(uint4*)l = *(const uint4*)g;
#endif
}
__device__ __forceinline__ void async_wait0() {
#if HAVE_ASYNC_LDS
#if __has_builtin(__builtin_amdgcn_s_wait_asynccnt)
  __builtin_amdgcn_s_wait_asynccnt(0);
#else
  asm volatile("s_wait_asynccnt 0x0" ::: "memory");
#endif
#endif
}

__device__ __forceinline__ unsigned short f2bf(float f) {
  unsigned u = __float_as_uint(f);
  u += 0x7FFFu + ((u >> 16) & 1u);           // round-to-nearest-even
  return (unsigned short)(u >> 16);
}
__device__ __forceinline__ unsigned pack2(float lo, float hi) {
  return (unsigned)f2bf(lo) | ((unsigned)f2bf(hi) << 16);
}
// Load one 16x32 bf16 WMMA fragment half-pattern from LDS:
// lane holds K = {k0..k0+7} and {k0+16..k0+23} (k0 = 8*(lane>>4)) -> two b128s.
__device__ __forceinline__ v16bf ldfrag(const unsigned short* p) {
  union { uint4 q[2]; v16bf v; } u;
  u.q[0] = *(const uint4*)p;
  u.q[1] = *(const uint4*)(p + 16);
  return u.v;
}
__device__ __forceinline__ v8f wmma_bf16(v16bf a, v16bf b, v8f c) {
  return __builtin_amdgcn_wmma_f32_16x16x32_bf16(false, a, false, b,
                                                 (short)0, c, false, false);
}

// ---------------------------------------------------------------------------
// GEMM: out[M,N] = A[M,K](bf16) @ W[N,K](f32->bf16).T + bias
// EPI: 0 = f32+bias, 1 = bf16+bias, 2 = relu->bf16+bias, 3 = f32 resid += .
// ---------------------------------------------------------------------------
#define BM 128
#define BN 128
#define BK 32
#define LDT 40   // padded LDS row stride (halves): conflict-free b128 frag reads

template <int EPI>
__global__ __launch_bounds__(256)
void gemm_bf16_kernel(const unsigned short* __restrict__ A,
                      const float* __restrict__ W,
                      const float* __restrict__ bias,
                      float* __restrict__ outF,
                      unsigned short* __restrict__ outH,
                      int M, int N, int K) {
  __shared__ __align__(16) unsigned short As[2][BM * LDT];
  __shared__ __align__(16) unsigned short Bs[2][BN * LDT];

  const int tid  = threadIdx.x;
  const int wave = tid >> 5;
  const int lane = tid & 31;
  const int l15  = lane & 15;
  const int lhv  = lane >> 4;
  const int nBase = blockIdx.x * BN;
  const int mBase = blockIdx.y * BM;
  const int wm = (wave >> 1) * 32;   // 0,32,64,96
  const int wn = (wave & 1) * 64;    // 0,64

  v8f acc[2][4] = {};
  uint2 bReg[4];
  const int KT = K / BK;

  // A tile 128x32 bf16: 2 x 16B per thread, async straight into LDS.
  auto issueA = [&](int kt, int buf) {
#pragma unroll
    for (int i = 0; i < 2; ++i) {
      int q = tid + i * 256;
      int r = q >> 2, c = q & 3;
      ld16_to_lds(A + (size_t)(mBase + r) * K + kt * BK + c * 8,
                  &As[buf][r * LDT + c * 8]);
    }
  };
  // W tile 128x32 f32 -> bf16 through VGPRs (conversion needed).
  auto fetchB = [&](int kt) {
#pragma unroll
    for (int i = 0; i < 4; ++i) {
      int q = tid + i * 256;
      int r = q >> 3, c = q & 7;
      float4 f = *(const float4*)(W + (size_t)(nBase + r) * K + kt * BK + c * 4);
      bReg[i].x = pack2(f.x, f.y);
      bReg[i].y = pack2(f.z, f.w);
    }
    if (kt + 2 < KT)                              // global_prefetch_b8
      __builtin_prefetch(W + (size_t)(nBase + (tid >> 1)) * K + (kt + 2) * BK, 0, 1);
  };
  auto commitB = [&](int buf) {
#pragma unroll
    for (int i = 0; i < 4; ++i) {
      int q = tid + i * 256;
      int r = q >> 3, c = q & 7;
      *(uint2*)&Bs[buf][r * LDT + c * 4] = bReg[i];
    }
  };

  issueA(0, 0);
  fetchB(0);
  for (int kt = 0; kt < KT; ++kt) {
    const int buf = kt & 1;
    commitB(buf);
    async_wait0();                  // A tile for `buf` resident in LDS
    __syncthreads();
    if (kt + 1 < KT) {
      issueA(kt + 1, buf ^ 1);      // overlap with compute below
      fetchB(kt + 1);
    }

    v16bf af[2], bf[4];
#pragma unroll
    for (int ti = 0; ti < 2; ++ti)
      af[ti] = ldfrag(&As[buf][(wm + ti * 16 + l15) * LDT + 8 * lhv]);
#pragma unroll
    for (int tj = 0; tj < 4; ++tj)
      bf[tj] = ldfrag(&Bs[buf][(wn + tj * 16 + l15) * LDT + 8 * lhv]);
#pragma unroll
    for (int ti = 0; ti < 2; ++ti)
#pragma unroll
      for (int tj = 0; tj < 4; ++tj)
        acc[ti][tj] = wmma_bf16(af[ti], bf[tj], acc[ti][tj]);
  }

  // Epilogue. C frag: VGPR g, lane l -> (m = 8*(l>>4)+g, n = l&15).
#pragma unroll
  for (int tj = 0; tj < 4; ++tj) {
    const int n = nBase + wn + tj * 16 + l15;
    const float bv = bias[n];
#pragma unroll
    for (int ti = 0; ti < 2; ++ti) {
#pragma unroll
      for (int g = 0; g < 8; ++g) {
        const int m = mBase + wm + ti * 16 + lhv * 8 + g;
        const size_t idx = (size_t)m * N + n;
        const float v = acc[ti][tj][g] + bv;
        if (EPI == 0) outF[idx] = v;
        else if (EPI == 3) outF[idx] += v;                 // residual add
        else if (EPI == 1) outH[idx] = f2bf(v);
        else outH[idx] = f2bf(fmaxf(v, 0.f));              // relu
      }
    }
  }
}

// ---------------------------------------------------------------------------
// Flash attention: one block = (b, h, 128 q rows). Each wave owns 16 q rows.
// qkv: bf16 [8192, 3072]  (q | k | v, each [.., h*64+hd]).
// ---------------------------------------------------------------------------
#define FST 72   // padded LDS stride for 64-wide tiles

__global__ __launch_bounds__(256)
void attn_kernel(const unsigned short* __restrict__ qkv,
                 const unsigned char* __restrict__ mask,
                 unsigned short* __restrict__ outO) {
  __shared__ __align__(16) unsigned short Qs[128][FST];
  __shared__ __align__(16) unsigned short Ks[64][FST];
  __shared__ __align__(16) unsigned short Vt[64][FST];       // [hd][key]
  __shared__ __align__(16) unsigned short Ps[8][16][FST];    // wave-private P

  const int tid = threadIdx.x, wave = tid >> 5, lane = tid & 31;
  const int l15 = lane & 15, lhv = lane >> 4;
  const int qb = blockIdx.x * 128;
  const int h = blockIdx.y, b = blockIdx.z;
  const size_t rowBase = (size_t)b * SEQ;
  const unsigned short* Qg = qkv + h * HEADD;
  const unsigned short* Kg = qkv + DMODEL + h * HEADD;
  const unsigned short* Vg = qkv + 2 * DMODEL + h * HEADD;

#pragma unroll
  for (int i = 0; i < 4; ++i) {                    // Q tile 128x64, async
    int q = tid + i * 256;
    int r = q >> 3, c = q & 7;
    ld16_to_lds(Qg + (rowBase + qb + r) * 3072 + c * 8, &Qs[r][c * 8]);
  }

  v8f o[4] = {};
  float mrow[8], lrow[8];
#pragma unroll
  for (int g = 0; g < 8; ++g) { mrow[g] = -1e30f; lrow[g] = 0.f; }

  for (int kb = 0; kb < SEQ; kb += 64) {
#pragma unroll
    for (int i = 0; i < 2; ++i) {                  // K tile 64x64, async
      int q = tid + i * 256;
      int r = q >> 3, c = q & 7;
      ld16_to_lds(Kg + (rowBase + kb + r) * 3072 + c * 8, &Ks[r][c * 8]);
    }
#pragma unroll
    for (int i = 0; i < 2; ++i) {                  // V tile, transposed store
      int q = tid + i * 256;
      int r = q >> 3, c = q & 7;
      union { uint4 q4; unsigned short s[8]; } u;
      u.q4 = *(const uint4*)(Vg + (rowBase + kb + r) * 3072 + c * 8);
#pragma unroll
      for (int e = 0; e < 8; ++e) Vt[c * 8 + e][r] = u.s[e];
    }
    async_wait0();          // covers K tile now, and the Q tile on first pass
    __syncthreads();

    // S strip = Q[16x64] @ K^T, K-dim = HD = 64 (2 WMMA steps)
    v8f s[4] = {};
#pragma unroll
    for (int ks = 0; ks < 2; ++ks) {
      v16bf aq = ldfrag(&Qs[wave * 16 + l15][ks * 32 + 8 * lhv]);
#pragma unroll
      for (int tj = 0; tj < 4; ++tj) {
        v16bf bk = ldfrag(&Ks[tj * 16 + l15][ks * 32 + 8 * lhv]);
        s[tj] = wmma_bf16(aq, bk, s[tj]);
      }
    }

    float mk[4];
#pragma unroll
    for (int tj = 0; tj < 4; ++tj) {
      const int key = kb + tj * 16 + l15;
      mk[tj] = mask[b * SEQ + key] ? -1e30f : 0.f;
    }

    // online softmax per owned row (g); row data spans the 16-lane half.
#pragma unroll
    for (int g = 0; g < 8; ++g) {
      float vm = -1e30f;
#pragma unroll
      for (int tj = 0; tj < 4; ++tj) {
        float sv = s[tj][g] * 0.125f + mk[tj];     // HD^-0.5 = 1/8
        s[tj][g] = sv;
        vm = fmaxf(vm, sv);
      }
#pragma unroll
      for (int off = 8; off >= 1; off >>= 1)
        vm = fmaxf(vm, __shfl_xor(vm, off, 32));
      const float mnew = fmaxf(mrow[g], vm);
      const float corr = __expf(mrow[g] - mnew);
      float rs = 0.f;
#pragma unroll
      for (int tj = 0; tj < 4; ++tj) {
        float p = __expf(s[tj][g] - mnew);
        s[tj][g] = p;
        rs += p;
      }
#pragma unroll
      for (int off = 8; off >= 1; off >>= 1) rs += __shfl_xor(rs, off, 32);
      lrow[g] = lrow[g] * corr + rs;
      mrow[g] = mnew;
#pragma unroll
      for (int tj = 0; tj < 4; ++tj) o[tj][g] *= corr;
    }

    // P -> LDS (wave private), re-enter as A fragment for O += P @ V
#pragma unroll
    for (int tj = 0; tj < 4; ++tj)
#pragma unroll
      for (int g = 0; g < 8; ++g)
        Ps[wave][g + 8 * lhv][tj * 16 + l15] = f2bf(s[tj][g]);

#pragma unroll
    for (int ks = 0; ks < 2; ++ks) {
      v16bf ap = ldfrag(&Ps[wave][l15][ks * 32 + 8 * lhv]);
#pragma unroll
      for (int tj = 0; tj < 4; ++tj) {
        v16bf bv = ldfrag(&Vt[tj * 16 + l15][ks * 32 + 8 * lhv]);
        o[tj] = wmma_bf16(ap, bv, o[tj]);
      }
    }
    __syncthreads();   // protect Ks/Vt before next key block
  }

#pragma unroll
  for (int g = 0; g < 8; ++g) {
    const float inv = 1.f / lrow[g];
    const int m = (int)rowBase + qb + wave * 16 + 8 * lhv + g;
#pragma unroll
    for (int tj = 0; tj < 4; ++tj) {
      const int col = h * HEADD + tj * 16 + l15;
      outO[(size_t)m * DMODEL + col] = f2bf(o[tj][g] * inv);
    }
  }
}

// ---------------------------------------------------------------------------
// LayerNorm over D=1024. One block per row; OUT_BF16 selects output buffer.
// ---------------------------------------------------------------------------
template <bool OUT_BF16>
__global__ __launch_bounds__(256)
void layernorm_kernel(const float* __restrict__ x,
                      const float* __restrict__ g, const float* __restrict__ bta,
                      unsigned short* __restrict__ outH,
                      float* __restrict__ outF) {
  const int row = blockIdx.x;
  const float* xr = x + (size_t)row * DMODEL;
  const int tid = threadIdx.x, wave = tid >> 5, lane = tid & 31;
  float v[4], s = 0.f, s2 = 0.f;
#pragma unroll
  for (int i = 0; i < 4; ++i) {
    v[i] = xr[tid + i * 256];
    s += v[i];
    s2 += v[i] * v[i];
  }
#pragma unroll
  for (int off = 16; off >= 1; off >>= 1) {
    s += __shfl_xor(s, off, 32);
    s2 += __shfl_xor(s2, off, 32);
  }
  __shared__ float red[2][8];
  if (lane == 0) { red[0][wave] = s; red[1][wave] = s2; }
  __syncthreads();
  s = 0.f; s2 = 0.f;
#pragma unroll
  for (int w = 0; w < 8; ++w) { s += red[0][w]; s2 += red[1][w]; }
  const float mean = s * (1.f / DMODEL);
  const float var = s2 * (1.f / DMODEL) - mean * mean;
  const float rstd = rsqrtf(var + 1e-5f);
#pragma unroll
  for (int i = 0; i < 4; ++i) {
    const int c = tid + i * 256;
    const float y = (v[i] - mean) * rstd * g[c] + bta[c];
    if (OUT_BF16) outH[(size_t)row * DMODEL + c] = f2bf(y);
    else          outF[(size_t)row * DMODEL + c] = y;
  }
}

// ---------------------------------------------------------------------------
// concat(x, pos_emb) -> bf16 [8192, 1280]
// ---------------------------------------------------------------------------
__global__ __launch_bounds__(256)
void concat_pe_kernel(const float* __restrict__ x,
                      unsigned short* __restrict__ out) {
  const int row = blockIdx.x;
  const int pos = row & (SEQ - 1);
#pragma unroll
  for (int i = 0; i < 5; ++i) {
    const int c = threadIdx.x + i * 256;
    float val;
    if (c < 1024) {
      val = x[(size_t)row * 1024 + c];
    } else {
      const int j = c - 1024;
      const int jj = j & 127;
      const float fr = __expf((float)jj * -0.0719557843f);  // -ln(1e4)/128
      const float ang = (float)pos * fr;
      val = (j < 128) ? __sinf(ang) : __cosf(ang);
    }
    out[(size_t)row * INCAT + c] = f2bf(val);
  }
}

// ---------------------------------------------------------------------------
extern "C" void kernel_launch(void* const* d_in, const int* in_sizes, int n_in,
                              void* d_out, int out_size, void* d_ws,
                              size_t ws_size, hipStream_t stream) {
  (void)in_sizes; (void)n_in; (void)out_size; (void)ws_size;
  const float* x      = (const float*)d_in[0];
  const unsigned char* mask = (const unsigned char*)d_in[1];
  const float* inp_w  = (const float*)d_in[2];
  const float* inp_b  = (const float*)d_in[3];
  const float* ln1_g  = (const float*)d_in[4];
  const float* ln1_b  = (const float*)d_in[5];
  const float* qkv_w  = (const float*)d_in[6];
  const float* qkv_b  = (const float*)d_in[7];
  const float* out_w  = (const float*)d_in[8];
  const float* out_b  = (const float*)d_in[9];
  const float* ln2_g  = (const float*)d_in[10];
  const float* ln2_b  = (const float*)d_in[11];
  const float* fc1_w  = (const float*)d_in[12];
  const float* fc1_b  = (const float*)d_in[13];
  const float* fc2_w  = (const float*)d_in[14];
  const float* fc2_b  = (const float*)d_in[15];
  const float* lnf_g  = (const float*)d_in[16];
  const float* lnf_b  = (const float*)d_in[17];

  char* ws = (char*)d_ws;
  const size_t MB = 1024ull * 1024ull;
  float*          h     = (float*)(ws);                    // 32 MB f32 hidden
  unsigned short* lnout = (unsigned short*)(ws + 32 * MB); // 16 MB bf16 LN out
  unsigned short* qkvb  = (unsigned short*)(ws + 48 * MB); // 48 MB bf16 qkv
  unsigned short* ffb   = (unsigned short*)(ws + 96 * MB); // 64 MB bf16 ff/xcat
  unsigned short* aob   = (unsigned short*)(ws + 160 * MB);// 16 MB bf16 attn out

  const dim3 blk(256);

  // input projection: h = concat(x, pe) @ inp_w.T + inp_b
  concat_pe_kernel<<<MTOT, blk, 0, stream>>>(x, ffb);
  gemm_bf16_kernel<0><<<dim3(DMODEL / BN, MTOT / BM), blk, 0, stream>>>(
      ffb, inp_w, inp_b, h, nullptr, MTOT, DMODEL, INCAT);

  for (int l = 0; l < NLAYER; ++l) {
    layernorm_kernel<true><<<MTOT, blk, 0, stream>>>(
        h, ln1_g + (size_t)l * DMODEL, ln1_b + (size_t)l * DMODEL, lnout, nullptr);
    gemm_bf16_kernel<1><<<dim3(3 * DMODEL / BN, MTOT / BM), blk, 0, stream>>>(
        lnout, qkv_w + (size_t)l * 3 * DMODEL * DMODEL,
        qkv_b + (size_t)l * 3 * DMODEL, nullptr, qkvb, MTOT, 3 * DMODEL, DMODEL);
    attn_kernel<<<dim3(SEQ / 128, NHEAD, 16), blk, 0, stream>>>(qkvb, mask, aob);
    gemm_bf16_kernel<3><<<dim3(DMODEL / BN, MTOT / BM), blk, 0, stream>>>(
        aob, out_w + (size_t)l * DMODEL * DMODEL, out_b + (size_t)l * DMODEL,
        h, nullptr, MTOT, DMODEL, DMODEL);
    layernorm_kernel<true><<<MTOT, blk, 0, stream>>>(
        h, ln2_g + (size_t)l * DMODEL, ln2_b + (size_t)l * DMODEL, lnout, nullptr);
    gemm_bf16_kernel<2><<<dim3(FFDIM / BN, MTOT / BM), blk, 0, stream>>>(
        lnout, fc1_w + (size_t)l * FFDIM * DMODEL, fc1_b + (size_t)l * FFDIM,
        nullptr, ffb, MTOT, FFDIM, DMODEL);
    gemm_bf16_kernel<3><<<dim3(DMODEL / BN, MTOT / BM), blk, 0, stream>>>(
        ffb, fc2_w + (size_t)l * DMODEL * FFDIM, fc2_b + (size_t)l * DMODEL,
        h, nullptr, MTOT, DMODEL, FFDIM);
  }

  layernorm_kernel<false><<<MTOT, blk, 0, stream>>>(
      h, lnf_g, lnf_b, nullptr, (float*)d_out);
}
